// XGAT_9225589752233
// MI455X (gfx1250) — matrix operand order repaired and checked by reading, not compiled, not run on hardware
//
#include <hip/hip_runtime.h>
#include <hip/hip_bf16.h>
#include <math.h>

#define DVEC 128          // feature dim per layer output
#define DTOT 768          // 128 * (DEPTH+1) * 2 GATs
#define GAMMA_F 3.0f

typedef __attribute__((ext_vector_type(16))) __bf16 v16bf;
typedef __attribute__((ext_vector_type(8)))  __bf16 v8bf;
typedef __attribute__((ext_vector_type(8)))  float  v8f;

// ---------- helpers ----------
__device__ __forceinline__ unsigned encf(float f) {
    unsigned u = __float_as_uint(f);
    return (u & 0x80000000u) ? ~u : (u | 0x80000000u);   // order-preserving uint key
}
__device__ __forceinline__ float decf(unsigned u) {
    u = (u & 0x80000000u) ? (u & 0x7fffffffu) : ~u;
    return __uint_as_float(u);
}
__device__ __forceinline__ float wredsum(float v) {
#pragma unroll
    for (int m = 16; m; m >>= 1) v += __shfl_xor(v, m, 32);
    return v;
}

// ---------- graph kernels (bandwidth-bound, wave-per-edge) ----------
__global__ void k_deg(const int* __restrict__ rows, int E, float* __restrict__ deg) {
    int e = blockIdx.x * blockDim.x + threadIdx.x;
    if (e < E) atomicAdd(deg + rows[e], 1.0f);
}

// out[row] += emb[col] / deg[row], D=128, one wave per edge (4 floats/lane)
__global__ void k_avg(const int* __restrict__ rows, const int* __restrict__ cols, int E,
                      const float* __restrict__ emb, const float* __restrict__ deg,
                      float* __restrict__ out) {
    int e = (blockIdx.x * blockDim.x + threadIdx.x) >> 5;
    int lane = threadIdx.x & 31;
    if (e >= E) return;
    int r = rows[e], c = cols[e];
    float inv = 1.0f / deg[r];
    const float4 v = *(const float4*)(emb + (size_t)c * DVEC + lane * 4);
    float* o = out + (size_t)r * DVEC + lane * 4;
    atomicAdd(o + 0, v.x * inv);
    atomicAdd(o + 1, v.y * inv);
    atomicAdd(o + 2, v.z * inv);
    atomicAdd(o + 3, v.w * inv);
}

// dst = tanh(src); also write into emb[:, colOff:colOff+128]
__global__ void k_tanh(const float* __restrict__ src, float* __restrict__ dstFeat,
                       float* __restrict__ emb, int colOff, int N) {
    int idx = blockIdx.x * blockDim.x + threadIdx.x;
    if (idx >= N * DVEC) return;
    int i = idx / DVEC, d = idx - i * DVEC;
    float t = tanhf(src[idx]);
    dstFeat[idx] = t;
    emb[(size_t)i * DTOT + colOff + d] = t;
}

// per-edge attention logit: tri = normalize(r_val * rel_emb[rind1]); att = tri . attn
__global__ void k_att1(const int* __restrict__ adjRow, const int* __restrict__ rind1,
                       const float* __restrict__ rval, const float* __restrict__ relEmb,
                       const float* __restrict__ attn, float* __restrict__ edgeAtt,
                       unsigned* __restrict__ nodeMx, int T) {
    int e = (blockIdx.x * blockDim.x + threadIdx.x) >> 5;
    int lane = threadIdx.x & 31;
    if (e >= T) return;
    int rr = rind1[e];
    float rv = rval[e];
    float4 v = *(const float4*)(relEmb + (size_t)rr * DVEC + lane * 4);
    v.x *= rv; v.y *= rv; v.z *= rv; v.w *= rv;
    float nn = wredsum(v.x * v.x + v.y * v.y + v.z * v.z + v.w * v.w);
    float inv = 1.0f / fmaxf(sqrtf(nn), 1e-12f);
    const float4 a = *(const float4*)(attn + lane * 4);
    float at = wredsum(v.x * a.x + v.y * a.y + v.z * a.z + v.w * a.w) * inv;
    if (lane == 0) {
        edgeAtt[e] = at;
        atomicMax(nodeMx + adjRow[e], encf(at));
    }
}

// e = exp(att - rowmax); edgeAtt = e; rowsum += e
__global__ void k_att2(const int* __restrict__ adjRow, float* __restrict__ edgeAtt,
                       const unsigned* __restrict__ nodeMx, float* __restrict__ nodeSm, int T) {
    int e = blockIdx.x * blockDim.x + threadIdx.x;
    if (e >= T) return;
    int r = adjRow[e];
    float ee = expf(edgeAtt[e] - decf(nodeMx[r]));
    edgeAtt[e] = ee;
    atomicAdd(nodeSm + r, ee);
}

// new[row] += (f - 2 (f.tri) tri) * softmax_att,  wave per edge
__global__ void k_scatter(const int* __restrict__ adjRow, const int* __restrict__ adjCol,
                          const int* __restrict__ rind1, const float* __restrict__ rval,
                          const float* __restrict__ relEmb, const float* __restrict__ edgeAtt,
                          const float* __restrict__ nodeSm, const float* __restrict__ feat,
                          float* __restrict__ out, int T) {
    int e = (blockIdx.x * blockDim.x + threadIdx.x) >> 5;
    int lane = threadIdx.x & 31;
    if (e >= T) return;
    int r = adjRow[e], c = adjCol[e], rr = rind1[e];
    float rv = rval[e];
    float4 v = *(const float4*)(relEmb + (size_t)rr * DVEC + lane * 4);
    v.x *= rv; v.y *= rv; v.z *= rv; v.w *= rv;
    float nn = wredsum(v.x * v.x + v.y * v.y + v.z * v.z + v.w * v.w);
    float inv = 1.0f / fmaxf(sqrtf(nn), 1e-12f);
    const float4 f = *(const float4*)(feat + (size_t)c * DVEC + lane * 4);
    float d = wredsum(f.x * v.x + f.y * v.y + f.z * v.z + f.w * v.w) * inv;
    float w = edgeAtt[e] / nodeSm[r];
    float s = 2.0f * d * inv;
    float* o = out + (size_t)r * DVEC + lane * 4;
    atomicAdd(o + 0, (f.x - s * v.x) * w);
    atomicAdd(o + 1, (f.y - s * v.y) * w);
    atomicAdd(o + 2, (f.z - s * v.z) * w);
    atomicAdd(o + 3, (f.w - s * v.w) * w);
}

// ---------- loss prep ----------
// sqe[n] = |emb[n]|^2 ; Eh/El = bf16 hi/lo split of emb (wave per node)
__global__ void k_embstats(const float* __restrict__ emb, __bf16* __restrict__ Eh,
                           __bf16* __restrict__ El, float* __restrict__ sqe, int N) {
    int n = (blockIdx.x * blockDim.x + threadIdx.x) >> 5;
    int lane = threadIdx.x & 31;
    if (n >= N) return;
    const float* row = emb + (size_t)n * DTOT;
    __bf16* eh = Eh + (size_t)n * DTOT;
    __bf16* el = El + (size_t)n * DTOT;
    float acc = 0.0f;
    for (int j = lane; j < DTOT; j += 32) {
        float x = row[j];
        acc += x * x;
        __bf16 h = (__bf16)x;
        eh[j] = h;
        el[j] = (__bf16)(x - (float)h);
    }
    acc = wredsum(acc);
    if (lane == 0) sqe[n] = acc;
}

// gather pair rows into contiguous A matrices (hi/lo), pos = |l-r|^2, sq norms
__global__ void k_pairs(const float* __restrict__ emb, const __bf16* __restrict__ Eh,
                        const __bf16* __restrict__ El, const int* __restrict__ tp,
                        __bf16* __restrict__ Alh, __bf16* __restrict__ All,
                        __bf16* __restrict__ Arh, __bf16* __restrict__ Arl,
                        float* __restrict__ pos, float* __restrict__ sqaL,
                        float* __restrict__ sqaR, const float* __restrict__ sqe, int B) {
    int i = (blockIdx.x * blockDim.x + threadIdx.x) >> 5;
    int lane = threadIdx.x & 31;
    if (i >= B) return;
    int l = tp[2 * i], r = tp[2 * i + 1];
    const float* el_ = emb + (size_t)l * DTOT;
    const float* er_ = emb + (size_t)r * DTOT;
    float acc = 0.0f;
    for (int j = lane; j < DTOT; j += 32) {
        float d = el_[j] - er_[j];
        acc += d * d;
        Alh[(size_t)i * DTOT + j] = Eh[(size_t)l * DTOT + j];
        All[(size_t)i * DTOT + j] = El[(size_t)l * DTOT + j];
        Arh[(size_t)i * DTOT + j] = Eh[(size_t)r * DTOT + j];
        Arl[(size_t)i * DTOT + j] = El[(size_t)r * DTOT + j];
    }
    acc = wredsum(acc);
    if (lane == 0) { pos[i] = acc; sqaL[i] = sqe[l]; sqaR[i] = sqe[r]; }
}

// ---------- WMMA core: 16x16 f32 tile of A . E^T via 3x bf16 split WMMAs ----------
__device__ __forceinline__ v8f mma_tile(const __bf16* __restrict__ Ah, const __bf16* __restrict__ Al,
                                        const __bf16* __restrict__ Eh, const __bf16* __restrict__ El,
                                        int rowBase, int colBase, int lane) {
    int half = lane >> 4, nl = lane & 15;
    const __bf16* ah = Ah + (size_t)(rowBase + nl) * DTOT + half * 16;
    const __bf16* al = Al + (size_t)(rowBase + nl) * DTOT + half * 16;
    const __bf16* bh = Eh + (size_t)(colBase + nl) * DTOT + half * 16;
    const __bf16* bl = El + (size_t)(colBase + nl) * DTOT + half * 16;
    v8f c = {};
#pragma unroll
    for (int k = 0; k < DTOT; k += 32) {
        v16bf avh = *(const v16bf*)(ah + k);
        v16bf avl = *(const v16bf*)(al + k);
        v16bf bvh = *(const v16bf*)(bh + k);
        v16bf bvl = *(const v16bf*)(bl + k);
        c = __builtin_amdgcn_wmma_f32_16x16x32_bf16(false, avh, false, bvh, (short)0, c, false, false);
        c = __builtin_amdgcn_wmma_f32_16x16x32_bf16(false, avh, false, bvl, (short)0, c, false, false);
        c = __builtin_amdgcn_wmma_f32_16x16x32_bf16(false, avl, false, bvh, (short)0, c, false, false);
    }
    return c;
}

// pass 1: per-row sum / sumsq / max of loss = (pos - sqdist + gamma) * mask
__global__ __launch_bounds__(256) void k_gemm_stats(
    const __bf16* __restrict__ Alh, const __bf16* __restrict__ All,
    const __bf16* __restrict__ Arh, const __bf16* __restrict__ Arl,
    const __bf16* __restrict__ Eh, const __bf16* __restrict__ El,
    const float* __restrict__ sqaL, const float* __restrict__ sqaR,
    const float* __restrict__ sqe, const float* __restrict__ pos,
    const int* __restrict__ tp,
    float* __restrict__ sumL, float* __restrict__ sqL, unsigned* __restrict__ mxL,
    float* __restrict__ sumR, float* __restrict__ sqR, unsigned* __restrict__ mxR,
    int nColTiles) {
    int side = blockIdx.y;
    const __bf16* Ah = side ? Arh : Alh;
    const __bf16* Al = side ? Arl : All;
    const float* sqa = side ? sqaR : sqaL;
    float* osum = side ? sumR : sumL;
    float* osq  = side ? sqR  : sqL;
    unsigned* omx = side ? mxR : mxL;
    int rowBase = blockIdx.x * 16;
    int lane = threadIdx.x & 31, wave = threadIdx.x >> 5;
    int half = lane >> 4, nl = lane & 15;
    float p8[8], q8[8]; int li8[8], ri8[8];
#pragma unroll
    for (int i = 0; i < 8; ++i) {
        int r = rowBase + i + 8 * half;
        p8[i] = pos[r]; q8[i] = sqa[r]; li8[i] = tp[2 * r]; ri8[i] = tp[2 * r + 1];
    }
    float s8[8], ss8[8], m8[8];
#pragma unroll
    for (int i = 0; i < 8; ++i) { s8[i] = 0.f; ss8[i] = 0.f; m8[i] = -3.4e38f; }
    for (int tile = wave; tile < nColTiles; tile += 8) {
        int colBase = tile * 16, col = colBase + nl;
        float se = sqe[col];
        v8f c = mma_tile(Ah, Al, Eh, El, rowBase, colBase, lane);
#pragma unroll
        for (int i = 0; i < 8; ++i) {
            float mf = 1.0f - (float)(col == li8[i]) - (float)(col == ri8[i]);
            float v = (p8[i] - (q8[i] + se - 2.0f * c[i]) + GAMMA_F) * mf;
            s8[i] += v; ss8[i] += v * v; m8[i] = fmaxf(m8[i], v);
        }
    }
#pragma unroll
    for (int m = 1; m < 16; m <<= 1) {
#pragma unroll
        for (int i = 0; i < 8; ++i) {
            s8[i]  += __shfl_xor(s8[i], m, 32);
            ss8[i] += __shfl_xor(ss8[i], m, 32);
            m8[i]   = fmaxf(m8[i], __shfl_xor(m8[i], m, 32));
        }
    }
    if (nl == 0) {
#pragma unroll
        for (int i = 0; i < 8; ++i) {
            int r = rowBase + i + 8 * half;
            atomicAdd(osum + r, s8[i]);
            atomicAdd(osq + r, ss8[i]);
            atomicMax(omx + r, encf(m8[i]));
        }
    }
}

// finalize row stats: mu, sigma, c=20/sd, shift
__global__ void k_rowfin(const float* __restrict__ sumL, const float* __restrict__ sqL,
                         const unsigned* __restrict__ mxL, float* __restrict__ cfL,
                         float* __restrict__ mxfL, float* __restrict__ shL,
                         const float* __restrict__ sumR, const float* __restrict__ sqR,
                         const unsigned* __restrict__ mxR, float* __restrict__ cfR,
                         float* __restrict__ mxfR, float* __restrict__ shR,
                         float Nf, int B) {
    int idx = blockIdx.x * blockDim.x + threadIdx.x;
    if (idx >= 2 * B) return;
    int side = idx >= B, r = idx - side * B;
    const float* sum = side ? sumR : sumL;
    const float* sq  = side ? sqR  : sqL;
    const unsigned* mxb = side ? mxR : mxL;
    float* cf = side ? cfR : cfL;
    float* mxf = side ? mxfR : mxfL;
    float* sh = side ? shR : shL;
    float mu = sum[r] / Nf;
    float var = sq[r] / Nf - mu * mu;
    float sd = sqrtf(fmaxf(var, 1e-30f));
    float mx = decf(mxb[r]);
    cf[r] = 20.0f / sd;
    mxf[r] = mx;
    sh[r] = (mx - mu) / sd * 20.0f + 8.0f;
}

// pass 2: per-row sum of exp(c * (v - max))  (all exponents <= 0)
__global__ __launch_bounds__(256) void k_gemm_exp(
    const __bf16* __restrict__ Alh, const __bf16* __restrict__ All,
    const __bf16* __restrict__ Arh, const __bf16* __restrict__ Arl,
    const __bf16* __restrict__ Eh, const __bf16* __restrict__ El,
    const float* __restrict__ sqaL, const float* __restrict__ sqaR,
    const float* __restrict__ sqe, const float* __restrict__ pos,
    const int* __restrict__ tp,
    const float* __restrict__ cfL, const float* __restrict__ mxfL,
    const float* __restrict__ cfR, const float* __restrict__ mxfR,
    float* __restrict__ seL, float* __restrict__ seR, int nColTiles) {
    int side = blockIdx.y;
    const __bf16* Ah = side ? Arh : Alh;
    const __bf16* Al = side ? Arl : All;
    const float* sqa = side ? sqaR : sqaL;
    const float* cf = side ? cfR : cfL;
    const float* mxf = side ? mxfR : mxfL;
    float* ose = side ? seR : seL;
    int rowBase = blockIdx.x * 16;
    int lane = threadIdx.x & 31, wave = threadIdx.x >> 5;
    int half = lane >> 4, nl = lane & 15;
    float p8[8], q8[8], c8[8], x8[8]; int li8[8], ri8[8];
#pragma unroll
    for (int i = 0; i < 8; ++i) {
        int r = rowBase + i + 8 * half;
        p8[i] = pos[r]; q8[i] = sqa[r]; li8[i] = tp[2 * r]; ri8[i] = tp[2 * r + 1];
        c8[i] = cf[r]; x8[i] = mxf[r];
    }
    float e8[8];
#pragma unroll
    for (int i = 0; i < 8; ++i) e8[i] = 0.0f;
    for (int tile = wave; tile < nColTiles; tile += 8) {
        int colBase = tile * 16, col = colBase + nl;
        float se = sqe[col];
        v8f c = mma_tile(Ah, Al, Eh, El, rowBase, colBase, lane);
#pragma unroll
        for (int i = 0; i < 8; ++i) {
            float mf = 1.0f - (float)(col == li8[i]) - (float)(col == ri8[i]);
            float v = (p8[i] - (q8[i] + se - 2.0f * c[i]) + GAMMA_F) * mf;
            e8[i] += expf(c8[i] * (v - x8[i]));
        }
    }
#pragma unroll
    for (int m = 1; m < 16; m <<= 1) {
#pragma unroll
        for (int i = 0; i < 8; ++i) e8[i] += __shfl_xor(e8[i], m, 32);
    }
    if (nl == 0) {
#pragma unroll
        for (int i = 0; i < 8; ++i) {
            int r = rowBase + i + 8 * half;
            atomicAdd(ose + r, e8[i]);
        }
    }
}

__global__ void k_final(const float* __restrict__ shL, const float* __restrict__ seL,
                        const float* __restrict__ shR, const float* __restrict__ seR,
                        int B, float* __restrict__ out) {
    __shared__ float sh[256];
    float acc = 0.0f;
    for (int i = threadIdx.x; i < B; i += 256)
        acc += shL[i] + logf(seL[i]) + shR[i] + logf(seR[i]);
    sh[threadIdx.x] = acc;
    __syncthreads();
    for (int s = 128; s; s >>= 1) {
        if (threadIdx.x < s) sh[threadIdx.x] += sh[threadIdx.x + s];
        __syncthreads();
    }
    if (threadIdx.x == 0) out[0] = sh[0] / (float)B;
}

// ---------- host side ----------
extern "C" void kernel_launch(void* const* d_in, const int* in_sizes, int n_in,
                              void* d_out, int out_size, void* d_ws, size_t ws_size,
                              hipStream_t stream) {
    (void)n_in; (void)out_size; (void)ws_size;
    const float* ent_emb = (const float*)d_in[0];
    const float* rel_emb = (const float*)d_in[1];
    const float* attn_e  = (const float*)d_in[2];
    const float* attn_r  = (const float*)d_in[3];
    const float* r_val   = (const float*)d_in[4];
    const int* adj       = (const int*)d_in[5];
    const int* r_index   = (const int*)d_in[6];
    const int* ent_adj   = (const int*)d_in[7];
    const int* rel_adj   = (const int*)d_in[8];
    const int* tp        = (const int*)d_in[9];

    const int N   = in_sizes[0] / DVEC;
    const int T   = in_sizes[5] / 2;
    const int NNZ = in_sizes[7] / 2;
    const int B   = in_sizes[9] / 2;

    const int* adjRow = adj;
    const int* adjCol = adj + T;
    const int* rind1  = r_index + T;

    // workspace carving (all 256B aligned)
    char* base = (char*)d_ws;
    size_t off = 0;
    auto alloc = [&](size_t bytes) -> void* {
        off = (off + 255) & ~(size_t)255;
        void* p = base + off;
        off += bytes;
        return p;
    };
    float*   emb   = (float*)alloc((size_t)N * DTOT * 4);
    __bf16*  Eh    = (__bf16*)alloc((size_t)N * DTOT * 2);
    __bf16*  El    = (__bf16*)alloc((size_t)N * DTOT * 2);
    float*   featA = (float*)alloc((size_t)N * DVEC * 4);
    float*   featB = (float*)alloc((size_t)N * DVEC * 4);
    float*   deg   = (float*)alloc((size_t)N * 4);
    unsigned* nodeMx = (unsigned*)alloc((size_t)N * 4);
    float*   nodeSm = (float*)alloc((size_t)N * 4);
    float*   edgeAtt = (float*)alloc((size_t)T * 4);
    __bf16*  Alh = (__bf16*)alloc((size_t)B * DTOT * 2);
    __bf16*  All = (__bf16*)alloc((size_t)B * DTOT * 2);
    __bf16*  Arh = (__bf16*)alloc((size_t)B * DTOT * 2);
    __bf16*  Arl = (__bf16*)alloc((size_t)B * DTOT * 2);
    float* pos  = (float*)alloc((size_t)B * 4);
    float* sqaL = (float*)alloc((size_t)B * 4);
    float* sqaR = (float*)alloc((size_t)B * 4);
    float* sqe  = (float*)alloc((size_t)N * 4);
    float* sumL = (float*)alloc((size_t)B * 4);
    float* sqL  = (float*)alloc((size_t)B * 4);
    unsigned* mxL = (unsigned*)alloc((size_t)B * 4);
    float* cfL  = (float*)alloc((size_t)B * 4);
    float* mxfL = (float*)alloc((size_t)B * 4);
    float* shL  = (float*)alloc((size_t)B * 4);
    float* seL  = (float*)alloc((size_t)B * 4);
    float* sumR = (float*)alloc((size_t)B * 4);
    float* sqR  = (float*)alloc((size_t)B * 4);
    unsigned* mxR = (unsigned*)alloc((size_t)B * 4);
    float* cfR  = (float*)alloc((size_t)B * 4);
    float* mxfR = (float*)alloc((size_t)B * 4);
    float* shR  = (float*)alloc((size_t)B * 4);
    float* seR  = (float*)alloc((size_t)B * 4);

    const int edgeWaves = (T + 7) / 8;
    const int edgeThreads = (T + 255) / 256;
    const int featThreads = (N * DVEC + 255) / 256;

    auto run_gat = [&](const float* attn, int colOff) {
        k_tanh<<<featThreads, 256, 0, stream>>>(featA, featA, emb, colOff, N);
        for (int l = 0; l < 2; ++l) {
            hipMemsetAsync(featB, 0, (size_t)N * DVEC * 4, stream);
            hipMemsetAsync(nodeMx, 0, (size_t)N * 4, stream);
            hipMemsetAsync(nodeSm, 0, (size_t)N * 4, stream);
            k_att1<<<edgeWaves, 256, 0, stream>>>(adjRow, rind1, r_val, rel_emb,
                                                  attn + l * DVEC, edgeAtt, nodeMx, T);
            k_att2<<<edgeThreads, 256, 0, stream>>>(adjRow, edgeAtt, nodeMx, nodeSm, T);
            k_scatter<<<edgeWaves, 256, 0, stream>>>(adjRow, adjCol, rind1, r_val, rel_emb,
                                                     edgeAtt, nodeSm, featA, featB, T);
            k_tanh<<<featThreads, 256, 0, stream>>>(featB, featA, emb,
                                                    colOff + (l + 1) * DVEC, N);
        }
    };

    // ---- entity branch: ent_feature = avg(ent_adj, ent_emb); GAT -> emb[:, 0:384]
    hipMemsetAsync(deg, 0, (size_t)N * 4, stream);
    hipMemsetAsync(featA, 0, (size_t)N * DVEC * 4, stream);
    k_deg<<<(NNZ + 255) / 256, 256, 0, stream>>>(ent_adj, NNZ, deg);
    k_avg<<<(NNZ + 7) / 8, 256, 0, stream>>>(ent_adj, ent_adj + NNZ, NNZ, ent_emb, deg, featA);
    run_gat(attn_e, 0);

    // ---- relation branch: rel_feature = avg(rel_adj, rel_emb); GAT -> emb[:, 384:768]
    hipMemsetAsync(deg, 0, (size_t)N * 4, stream);
    hipMemsetAsync(featA, 0, (size_t)N * DVEC * 4, stream);
    k_deg<<<(NNZ + 255) / 256, 256, 0, stream>>>(rel_adj, NNZ, deg);
    k_avg<<<(NNZ + 7) / 8, 256, 0, stream>>>(rel_adj, rel_adj + NNZ, NNZ, rel_emb, deg, featA);
    run_gat(attn_r, 384);

    // ---- alignment loss
    k_embstats<<<(N + 7) / 8, 256, 0, stream>>>(emb, Eh, El, sqe, N);
    k_pairs<<<(B + 7) / 8, 256, 0, stream>>>(emb, Eh, El, tp, Alh, All, Arh, Arl,
                                             pos, sqaL, sqaR, sqe, B);
    hipMemsetAsync(sumL, 0, (size_t)B * 4, stream);
    hipMemsetAsync(sqL, 0, (size_t)B * 4, stream);
    hipMemsetAsync(mxL, 0, (size_t)B * 4, stream);
    hipMemsetAsync(sumR, 0, (size_t)B * 4, stream);
    hipMemsetAsync(sqR, 0, (size_t)B * 4, stream);
    hipMemsetAsync(mxR, 0, (size_t)B * 4, stream);

    int nColTiles = N / 16;  // 30000/16 = 1875
    dim3 ggrid(B / 16, 2);
    k_gemm_stats<<<ggrid, 256, 0, stream>>>(Alh, All, Arh, Arl, Eh, El, sqaL, sqaR,
                                            sqe, pos, tp, sumL, sqL, mxL, sumR, sqR, mxR,
                                            nColTiles);
    k_rowfin<<<(2 * B + 255) / 256, 256, 0, stream>>>(sumL, sqL, mxL, cfL, mxfL, shL,
                                                      sumR, sqR, mxR, cfR, mxfR, shR,
                                                      (float)N, B);
    hipMemsetAsync(seL, 0, (size_t)B * 4, stream);
    hipMemsetAsync(seR, 0, (size_t)B * 4, stream);
    k_gemm_exp<<<ggrid, 256, 0, stream>>>(Alh, All, Arh, Arl, Eh, El, sqaL, sqaR,
                                          sqe, pos, tp, cfL, mxfL, cfR, mxfR,
                                          seL, seR, nColTiles);
    k_final<<<1, 256, 0, stream>>>(shL, seL, shR, seR, B, (float*)d_out);
}
